// OfficialStaticNerf_30709016166427
// MI455X (gfx1250) — matrix operand order repaired and checked by reading, not compiled, not run on hardware
//
#include <hip/hip_runtime.h>
#include <hip/hip_bf16.h>

typedef __attribute__((ext_vector_type(16))) _Float16 v16h;
typedef __attribute__((ext_vector_type(8)))  _Float16 v8h;
typedef __attribute__((ext_vector_type(8)))  float    v8f;
typedef __attribute__((ext_vector_type(4)))  unsigned int v4u;
typedef __attribute__((ext_vector_type(8)))  int          v8i;
typedef __attribute__((ext_vector_type(4)))  int          v4i;

#define WROW 40   // staged weight row stride in halves: 20 DWORDs = 4*5 mod 64
                  // -> conflict-free b128 B-fragment reads (banks 4*perm(16))

// ---------------------------------------------------------------------------
// Weight conversion: fp32 W (K x N row-major) -> f16 Wt (N x Kpad row-major,
// zero-padded rows K..Kpad-1). B-fragment = halves k 8h..8h+7 and 16+8h..23+8h per lane.
// ---------------------------------------------------------------------------
__global__ void convert_wt_kernel(const float* __restrict__ W,
                                  _Float16* __restrict__ Wt,
                                  int K, int N, int Kpad) {
  int idx = blockIdx.x * blockDim.x + threadIdx.x;
  int total = N * Kpad;
  if (idx >= total) return;
  int n = idx / Kpad;
  int k = idx - n * Kpad;
  float v = (k < K) ? W[k * N + n] : 0.0f;
  Wt[idx] = (_Float16)v;
}

__device__ inline v16h cat16(v8h lo, v8h hi) {
  return __builtin_shufflevector(lo, hi, 0, 1, 2, 3, 4, 5, 6, 7,
                                 8, 9, 10, 11, 12, 13, 14, 15);
}

// A-fragment: 16-bit A 16x32 layout -> two 16B chunks per lane (2x ds_load_b128)
__device__ inline v16h load_a_frag(const _Float16* base, int hk) {
  v8h lo = *(const v8h*)(base + hk * 8);
  v8h hi = *(const v8h*)(base + 16 + hk * 8);
  return cat16(lo, hi);
}

__device__ inline v16h load_b_frag(const _Float16* wb) {
  v8h lo = *(const v8h*)(wb);
  v8h hi = *(const v8h*)(wb + 16);
  return cat16(lo, hi);
}

// ---------------------------------------------------------------------------
// TDM staging: DMA Nrows x 32 halves (row stride Kpad halves) from the global
// weight matrix into LDS with 4-DWORD padding per 16-DWORD row -> staged row
// stride WROW halves. Wave 0 issues; TENSORcnt tracks; barrier publishes.
// ---------------------------------------------------------------------------
__device__ inline void stage_group(const _Float16* gsrc, int KpadHalves,
                                   int Nrows, _Float16* wlds) {
#if __has_builtin(__builtin_amdgcn_tensor_load_to_lds)
  if (threadIdx.x < 32) {
    unsigned int lds_off = (unsigned int)(uintptr_t)wlds;   // addr[31:0]
    unsigned long long ga = (unsigned long long)(uintptr_t)gsrc;
    // D# group 0: count=1, lds_addr, global_addr[56:0], type=2
    v4u g0 = { 1u,
               lds_off,
               (unsigned int)ga,
               (unsigned int)((ga >> 32) & 0x01FFFFFFu) | (2u << 30) };
    // D# group 1: data_size=2B; pad_enable, pad_interval=16 DW (code 3),
    // pad_amount=4 DW (code 3); huge tensor dims (no OOB; addr = tile start);
    // tile_dim0=32, tile_dim1=Nrows; stride0=Kpad elements.
    v8i g1 = { (int)0x06D10000u,                      // ds=1|pad_en|intv=3|amt=3
               (int)0xFFFF0000u,                      // tdim0[15:0]<<16
               (int)0xFFFF7FFFu,                      // tdim0 hi | tdim1 lo<<16
               (int)(0x7FFFu | (32u << 16)),          // tdim1 hi | tile_dim0=32
               (int)(unsigned int)Nrows,              // tile_dim1 | tile_dim2=0
               (int)(unsigned int)KpadHalves,         // tensor_dim0_stride
               0, 0 };
    v4i gz = { 0, 0, 0, 0 };
#if __clang_major__ >= 23
    v8i gz8 = { 0, 0, 0, 0, 0, 0, 0, 0 };
    __builtin_amdgcn_tensor_load_to_lds(g0, g1, gz, gz, gz8, 0);
#else
    __builtin_amdgcn_tensor_load_to_lds(g0, g1, gz, gz, 0);
#endif
  }
#else
  for (int i = threadIdx.x; i < Nrows * 4; i += (int)blockDim.x) {
    int r = i >> 2, cv = i & 3;
    *(v8h*)(wlds + r * WROW + cv * 8) =
        *(const v8h*)(gsrc + (size_t)r * KpadHalves + cv * 8);
  }
#endif
}

__device__ inline void stage_wait() {
#if __has_builtin(__builtin_amdgcn_tensor_load_to_lds)
  if (threadIdx.x < 32) __builtin_amdgcn_s_wait_tensorcnt(0);
#endif
}

// ---------------------------------------------------------------------------
// Accumulate one 32-wide K-group into NT persistent 16x16 f32 tiles.
// B fragments are software-pipelined 2 deep so two ds_load_b128 pairs stay
// in flight behind every WMMA instead of load->wait(0)->wmma serialization.
// ---------------------------------------------------------------------------
template <int NT>
__device__ inline void accum_group(v8f (&acc)[NT],
                                   const _Float16* src, int ldSrc, int kcBase,
                                   const _Float16* wlds) {
  const int lane = threadIdx.x & 31;
  const int n = lane & 15, hk = lane >> 4, wave = threadIdx.x >> 5;
  const int row = wave * 16 + n;
  v16h A = load_a_frag(src + row * ldSrc + kcBase * 32, hk);
  const _Float16* wbase = wlds + n * WROW + hk * 8;   // per-lane base; per-tile
                                                      // offset = nt*16*WROW
  v16h b0 = load_b_frag(wbase);
  v16h b1 = load_b_frag(wbase + 16 * WROW);
#pragma unroll
  for (int nt = 0; nt < NT; ++nt) {
    v16h bn = b1;
    if (nt + 2 < NT) bn = load_b_frag(wbase + (nt + 2) * 16 * WROW);
    acc[nt] = __builtin_amdgcn_wmma_f32_16x16x32_f16(
        false, A, false, b0, (short)0, acc[nt], false, false);
    b0 = b1;
    b1 = bn;
  }
}

template <int NT, bool RELU>
__device__ inline void finalize_store(v8f (&acc)[NT], const float* __restrict__ bias,
                                      _Float16* outB, int ldOut) {
  const int lane = threadIdx.x & 31;
  const int n = lane & 15, hk = lane >> 4, wave = threadIdx.x >> 5;
#pragma unroll
  for (int nt = 0; nt < NT; ++nt) {
    float bn = bias[nt * 16 + n];
#pragma unroll
    for (int r = 0; r < 8; ++r) {
      float v = acc[nt][r] + bn;
      if (RELU) v = fmaxf(v, 0.0f);
      outB[(wave * 16 + r + 8 * hk) * ldOut + nt * 16 + n] = (_Float16)v;
    }
  }
}

// ---------------------------------------------------------------------------
// One layer: persistent accumulators; uniform 32-wide K-groups pipelined
// through two staged weight buffers (TDM for g+1 overlaps WMMAs on g).
// Groups 0..nMain-1 read A from mainSrc (kcBase=g); the remaining nEnc groups
// read from encSrc (kcBase=g-nMain). In-place safe: each wave reads only its
// own rows and stores only after all groups.
// ---------------------------------------------------------------------------
template <int NT, bool RELU>
__device__ inline void run_layer(const _Float16* __restrict__ Wt, int Kpad,
                                 const _Float16* mainSrc, int mainLd, int nMain,
                                 const _Float16* encSrc, int encLd, int nEnc,
                                 const float* __restrict__ bias,
                                 _Float16* outB, int ldOut,
                                 _Float16* wb0, _Float16* wb1) {
  v8f acc[NT] = {};
  __builtin_prefetch(Wt, 0, 1);   // global_prefetch_b8: warm L2 ahead of TDM
  __syncthreads();                // weight buffers free (prev layer done)
  const int nG = nMain + nEnc;
  stage_group(Wt, Kpad, NT * 16, wb0);
  for (int g = 0; g < nG; ++g) {
    stage_wait();                 // wave0: TENSORcnt==0 for group g
    __syncthreads();              // group g visible to all waves
    if (g + 1 < nG)               // overlap next DMA with this group's WMMAs
      stage_group(Wt + (g + 1) * 32, Kpad, NT * 16, (g & 1) ? wb0 : wb1);
    const _Float16* src = (g < nMain) ? mainSrc : encSrc;
    const int ldSrc     = (g < nMain) ? mainLd : encLd;
    const int kcBase    = (g < nMain) ? g : (g - nMain);
    accum_group<NT>(acc, src, ldSrc, kcBase, (g & 1) ? wb1 : wb0);
  }
  finalize_store<NT, RELU>(acc, bias, outB, ldOut);
}

// ---------------------------------------------------------------------------
// Fused NeRF MLP: 128 points per 256-thread block (8 waves x 16 rows).
// ---------------------------------------------------------------------------
__global__ __launch_bounds__(256, 2)
void nerf_fused_kernel(const float* __restrict__ p,
                       const float* __restrict__ ray_d,
                       const _Float16* __restrict__ ws,
                       const float* __restrict__ b00, const float* __restrict__ b01,
                       const float* __restrict__ b02, const float* __restrict__ b03,
                       const float* __restrict__ b10, const float* __restrict__ b11,
                       const float* __restrict__ b12, const float* __restrict__ b13,
                       const float* __restrict__ Wd,  const float* __restrict__ bd,
                       const float* __restrict__ bf,
                       const float* __restrict__ br,
                       const float* __restrict__ Wc,  const float* __restrict__ bc,
                       float* __restrict__ out) {
  __shared__ __align__(32) _Float16 buf[128 * 256];   // 64 KB activations
  __shared__ __align__(32) _Float16 penc[128 * 64];   // 16 KB pos-enc
  __shared__ __align__(32) _Float16 denc[128 * 32];   //  8 KB dir-enc
  __shared__ __align__(32) _Float16 wb0[256 * WROW];  // 20 KB weight stage 0
  __shared__ __align__(32) _Float16 wb1[256 * WROW];  // 20 KB weight stage 1
  __shared__ float ps[128 * 3];
  __shared__ float rs[128 * 3];

  const int tid  = threadIdx.x;
  const int base = blockIdx.x * 128;

  for (int i = tid; i < 128 * 3; i += 256) {
    ps[i] = p[base * 3 + i];
    rs[i] = ray_d[base * 3 + i];
  }
  __syncthreads();

  for (int j = tid; j < 128 * 64; j += 256) {         // pos-enc, col63 pad
    int pt = j >> 6, d = j & 63;
    float v;
    if (d < 3) v = ps[pt * 3 + d];
    else if (d < 63) {
      int q = d - 3, lvl = q / 6, r = q % 6, c = r % 3;
      float t = ps[pt * 3 + c] * (float)(1 << lvl);
      v = (r < 3) ? __sinf(t) : __cosf(t);
    } else v = 0.0f;
    penc[j] = (_Float16)v;
  }
  for (int j = tid; j < 128 * 32; j += 256) {         // dir-enc, cols 27..31 pad
    int pt = j >> 5, d = j & 31;
    float v;
    if (d < 3) v = rs[pt * 3 + d];
    else if (d < 27) {
      int q = d - 3, lvl = q / 6, r = q % 6, c = r % 3;
      float t = rs[pt * 3 + c] * (float)(1 << lvl);
      v = (r < 3) ? __sinf(t) : __cosf(t);
    } else v = 0.0f;
    denc[j] = (_Float16)v;
  }

  const _Float16* Wt00 = ws + 0;        // 256 x 64
  const _Float16* Wt01 = ws + 16384;    // 256 x 256
  const _Float16* Wt02 = ws + 81920;
  const _Float16* Wt03 = ws + 147456;
  const _Float16* Wt10 = ws + 212992;   // 256 x 320 (skip)
  const _Float16* Wt11 = ws + 294912;
  const _Float16* Wt12 = ws + 360448;
  const _Float16* Wt13 = ws + 425984;
  const _Float16* Wtf  = ws + 491520;   // 256 x 256
  const _Float16* Wtr  = ws + 557056;   // 128 x 288

  run_layer<16, true>(Wt00, 64,  penc, 64, 2, nullptr, 0, 0, b00, buf, 256, wb0, wb1);
  run_layer<16, true>(Wt01, 256, buf, 256, 8, nullptr, 0, 0, b01, buf, 256, wb0, wb1);
  run_layer<16, true>(Wt02, 256, buf, 256, 8, nullptr, 0, 0, b02, buf, 256, wb0, wb1);
  run_layer<16, true>(Wt03, 256, buf, 256, 8, nullptr, 0, 0, b03, buf, 256, wb0, wb1);
  run_layer<16, true>(Wt10, 320, buf, 256, 8, penc, 64, 2, b10, buf, 256, wb0, wb1);
  run_layer<16, true>(Wt11, 256, buf, 256, 8, nullptr, 0, 0, b11, buf, 256, wb0, wb1);
  run_layer<16, true>(Wt12, 256, buf, 256, 8, nullptr, 0, 0, b12, buf, 256, wb0, wb1);
  run_layer<16, true>(Wt13, 256, buf, 256, 8, nullptr, 0, 0, b13, buf, 256, wb0, wb1);
  __syncthreads();  // density reads rows produced by all waves

  if (tid < 128) {  // density = softplus(x @ Wd + bd)
    float s = bd[0];
#pragma unroll 4
    for (int k = 0; k < 256; ++k) s += (float)buf[tid * 256 + k] * Wd[k];
    float sp = (s > 20.0f) ? s : __logf(1.0f + __expf(s));
    out[(size_t)(base + tid) * 4 + 3] = sp;
  }

  // feat = x @ Wf + bf (no relu), in-place; density readers are protected by
  // the in-layer barriers that precede any finalize store.
  run_layer<16, false>(Wtf, 256, buf, 256, 8, nullptr, 0, 0, bf, buf, 256, wb0, wb1);
  // h = relu([feat, dir_enc] @ Wr + br) -> cols 0..127, in-place
  run_layer<8, true>(Wtr, 288, buf, 256, 8, denc, 32, 1, br, buf, 256, wb0, wb1);
  __syncthreads();  // rgb reads rows produced by all waves

  for (int idx = tid; idx < 128 * 3; idx += 256) {  // rgb = sigmoid(h@Wc+bc)
    int pt = idx / 3, c = idx - pt * 3;
    float s = bc[c];
#pragma unroll 4
    for (int k = 0; k < 128; ++k) s += (float)buf[pt * 256 + k] * Wc[k * 3 + c];
    float sg = 1.0f / (1.0f + __expf(-s));
    out[(size_t)(base + pt) * 4 + c] = sg;
  }
}

// ---------------------------------------------------------------------------
extern "C" void kernel_launch(void* const* d_in, const int* in_sizes, int n_in,
                              void* d_out, int out_size, void* d_ws, size_t ws_size,
                              hipStream_t stream) {
  const float* p   = (const float*)d_in[0];
  const float* rd  = (const float*)d_in[1];
  const float* W00 = (const float*)d_in[2];  const float* b00 = (const float*)d_in[3];
  const float* W01 = (const float*)d_in[4];  const float* b01 = (const float*)d_in[5];
  const float* W02 = (const float*)d_in[6];  const float* b02 = (const float*)d_in[7];
  const float* W03 = (const float*)d_in[8];  const float* b03 = (const float*)d_in[9];
  const float* W10 = (const float*)d_in[10]; const float* b10 = (const float*)d_in[11];
  const float* W11 = (const float*)d_in[12]; const float* b11 = (const float*)d_in[13];
  const float* W12 = (const float*)d_in[14]; const float* b12 = (const float*)d_in[15];
  const float* W13 = (const float*)d_in[16]; const float* b13 = (const float*)d_in[17];
  const float* Wd  = (const float*)d_in[18]; const float* bd  = (const float*)d_in[19];
  const float* Wf  = (const float*)d_in[20]; const float* bf  = (const float*)d_in[21];
  const float* Wr  = (const float*)d_in[22]; const float* br  = (const float*)d_in[23];
  const float* Wc  = (const float*)d_in[24]; const float* bc  = (const float*)d_in[25];

  _Float16* ws = (_Float16*)d_ws;

  struct Cvt { const float* W; int K, N, Kpad; size_t off; };
  const Cvt cv[10] = {
      {W00,  63, 256,  64,      0},
      {W01, 256, 256, 256,  16384},
      {W02, 256, 256, 256,  81920},
      {W03, 256, 256, 256, 147456},
      {W10, 319, 256, 320, 212992},
      {W11, 256, 256, 256, 294912},
      {W12, 256, 256, 256, 360448},
      {W13, 256, 256, 256, 425984},
      {Wf,  256, 256, 256, 491520},
      {Wr,  283, 128, 288, 557056},
  };
  for (int i = 0; i < 10; ++i) {
    int total = cv[i].N * cv[i].Kpad;
    convert_wt_kernel<<<(total + 255) / 256, 256, 0, stream>>>(
        cv[i].W, ws + cv[i].off, cv[i].K, cv[i].N, cv[i].Kpad);
  }

  const int n_pts   = in_sizes[0] / 3;   // 524288
  const int nblocks = n_pts / 128;       // 4096 blocks, 8 waves each
  nerf_fused_kernel<<<nblocks, 256, 0, stream>>>(
      p, rd, ws,
      b00, b01, b02, b03, b10, b11, b12, b13,
      Wd, bd, bf, br, Wc, bc, (float*)d_out);
}
